// GCN_34205119545741
// MI455X (gfx1250) — matrix operand order repaired and checked by reading, not compiled, not run on hardware
//
#include <hip/hip_runtime.h>
#include <hip/hip_bf16.h>
#include <math.h>

// Problem constants (match reference)
#define BT   256     // B*T
#define NN   512     // boxes per frame
#define FG   128     // NFG
#define FR   64      // NFR
#define THR2 256.0f  // (0.2*80)^2

typedef __attribute__((ext_vector_type(16))) __bf16 v16bf;
typedef __attribute__((ext_vector_type(8)))  float  v8f;

__device__ __forceinline__ v8f wmma_bf16(v16bf a, v16bf b, v8f c) {
  return __builtin_amdgcn_wmma_f32_16x16x32_bf16(
      /*neg_a=*/false, a, /*neg_b=*/false, b,
      /*c_mod=*/(short)0, c, /*reuse_a=*/false, /*reuse_b=*/false);
}

__device__ __forceinline__ v8f zero8() {
  v8f z = {0.f,0.f,0.f,0.f,0.f,0.f,0.f,0.f};
  return z;
}

// ---- fragment loaders ---------------------------------------------------
// A-matrix 16x32 bf16: lane<16 -> row=lane, K[k0..k0+15];
// lane>=16 -> row=lane-16, K[k0+16..k0+31].
__device__ __forceinline__ v16bf afrag_from_f32(const float* __restrict__ src,
                                                int ld, long row0, int k0, int lane) {
  const int lr = lane & 15, hi = lane >> 4;
  const float* p = src + (row0 + lr) * (long)ld + k0 + hi * 16;
  float4 f0 = *(const float4*)(p);
  float4 f1 = *(const float4*)(p + 4);
  float4 f2 = *(const float4*)(p + 8);
  float4 f3 = *(const float4*)(p + 12);
  v16bf a;
  a[0]=(__bf16)f0.x; a[1]=(__bf16)f0.y; a[2]=(__bf16)f0.z; a[3]=(__bf16)f0.w;
  a[4]=(__bf16)f1.x; a[5]=(__bf16)f1.y; a[6]=(__bf16)f1.z; a[7]=(__bf16)f1.w;
  a[8]=(__bf16)f2.x; a[9]=(__bf16)f2.y; a[10]=(__bf16)f2.z; a[11]=(__bf16)f2.w;
  a[12]=(__bf16)f3.x; a[13]=(__bf16)f3.y; a[14]=(__bf16)f3.z; a[15]=(__bf16)f3.w;
  return a;
}

__device__ __forceinline__ v16bf afrag_from_bf16(const __bf16* __restrict__ src,
                                                 int ld, long row0, int k0, int lane) {
  const int lr = lane & 15, hi = lane >> 4;
  const __bf16* p = src + (row0 + lr) * (long)ld + k0 + hi * 16;
  union { v16bf v; uint4 q[2]; } u;
  u.q[0] = *(const uint4*)(p);
  u.q[1] = *(const uint4*)(p + 8);
  return u.v;
}

// B-matrix 32x16 bf16, B[k][n] = src[n][k]: lane<16 -> col=lane, K[k0..k0+15];
// lane>=16 -> col=lane-16, K[k0+16..k0+31].
__device__ __forceinline__ v16bf bfrag_from_f32(const float* __restrict__ src,
                                                int ld, long n0, int k0, int lane) {
  const int lr = lane & 15, hi = lane >> 4;
  const float* p = src + (n0 + lr) * (long)ld + k0 + hi * 16;
  float4 f0 = *(const float4*)(p);
  float4 f1 = *(const float4*)(p + 4);
  float4 f2 = *(const float4*)(p + 8);
  float4 f3 = *(const float4*)(p + 12);
  v16bf b;
  b[0]=(__bf16)f0.x; b[1]=(__bf16)f0.y; b[2]=(__bf16)f0.z; b[3]=(__bf16)f0.w;
  b[4]=(__bf16)f1.x; b[5]=(__bf16)f1.y; b[6]=(__bf16)f1.z; b[7]=(__bf16)f1.w;
  b[8]=(__bf16)f2.x; b[9]=(__bf16)f2.y; b[10]=(__bf16)f2.z; b[11]=(__bf16)f2.w;
  b[12]=(__bf16)f3.x; b[13]=(__bf16)f3.y; b[14]=(__bf16)f3.z; b[15]=(__bf16)f3.w;
  return b;
}

__device__ __forceinline__ v16bf bfrag_from_lds(const __bf16* p) {
  // p -> 16 contiguous bf16 (32B)
  union { v16bf v; uint4 q[2]; } u;
  u.q[0] = *(const uint4*)(p);
  u.q[1] = *(const uint4*)(p + 8);
  return u.v;
}

// ---- kernel 1: theta/phi projections (bf16 out to ws) -------------------
__global__ __launch_bounds__(128) void gcn_proj_kernel(
    const float* __restrict__ X, const float* __restrict__ tw,
    const float* __restrict__ tb, const float* __restrict__ pw,
    const float* __restrict__ pb, __bf16* __restrict__ theta,
    __bf16* __restrict__ phi) {
  const int lane = threadIdx.x & 31, wave = threadIdx.x >> 5;
  const int lr = lane & 15, hi = lane >> 4;
  const long row0 = (long)blockIdx.x * 64 + wave * 16;  // into [BT*NN]

  v16bf a[4];
#pragma unroll
  for (int kc = 0; kc < 4; ++kc) a[kc] = afrag_from_f32(X, FG, row0, kc * 32, lane);

#pragma unroll
  for (int nt = 0; nt < 4; ++nt) {
    const int n0 = nt * 16;
    v8f at = zero8(), ap = zero8();
#pragma unroll
    for (int kc = 0; kc < 4; ++kc) {
      at = wmma_bf16(a[kc], bfrag_from_f32(tw, FG, n0, kc * 32, lane), at);
      ap = wmma_bf16(a[kc], bfrag_from_f32(pw, FG, n0, kc * 32, lane), ap);
    }
    const float bt_ = tb[n0 + lr];
    const float bp_ = pb[n0 + lr];
#pragma unroll
    for (int v = 0; v < 8; ++v) {
      const long r = row0 + v + 8 * hi;
      theta[r * FR + n0 + lr] = (__bf16)(at[v] + bt_);
      phi[r * FR + n0 + lr]   = (__bf16)(ap[v] + bp_);
    }
  }
}

// ---- kernel 2 (fused): masked logits + row softmax ----------------------
// One block = one frame's 64-row strip x all 512 cols. Dynamic LDS:
//   cx/cy centers (4KB) | row max/inv (512B) | phi tile bf16 (64KB) |
//   simbuf f32 [4 waves][16 rows][512 cols] (128KB)
#define SM_CX     0
#define SM_CY     (SM_CX + NN * 4)
#define SM_ROWM   (SM_CY + NN * 4)
#define SM_ROWI   (SM_ROWM + 64 * 4)
#define SM_PHI    (SM_ROWI + 64 * 4)
#define SM_SIM    (SM_PHI + NN * FR * 2)
#define SM_BYTES  (SM_SIM + 4 * 16 * NN * 4)

__global__ __launch_bounds__(128) void gcn_sim_softmax_kernel(
    const __bf16* __restrict__ theta, const __bf16* __restrict__ phi,
    const float* __restrict__ boxes, float* __restrict__ rel) {
  extern __shared__ char smem[];
  float*  cx     = (float*)(smem + SM_CX);
  float*  cy     = (float*)(smem + SM_CY);
  float*  rowm   = (float*)(smem + SM_ROWM);
  float*  rowinv = (float*)(smem + SM_ROWI);
  __bf16* phils  = (__bf16*)(smem + SM_PHI);
  float*  simbuf = (float*)(smem + SM_SIM);

  const int f = blockIdx.x >> 3;
  const int tile = blockIdx.x & 7;

  // stage centers
  for (int i = threadIdx.x; i < NN; i += 128) {
    float4 bx = *(const float4*)(boxes + ((long)f * NN + i) * 4);
    cx[i] = (bx.x + bx.z) * 0.5f;
    cy[i] = (bx.y + bx.w) * 0.5f;
  }
  // stage phi tile for the whole frame (bf16, 512x64)
  {
    const uint4* src = (const uint4*)(phi + (long)f * NN * FR);
    uint4* dst = (uint4*)phils;
    for (int i = threadIdx.x; i < NN * FR / 8; i += 128) dst[i] = src[i];
  }
  __syncthreads();

  const int lane = threadIdx.x & 31, wave = threadIdx.x >> 5;
  const int lr = lane & 15, hi = lane >> 4;
  const int rowbase = tile * 64 + wave * 16;
  const long gr = (long)f * NN + rowbase;
  float* mysim = simbuf + wave * 16 * NN;

  const v16bf a0 = afrag_from_bf16(theta, FR, gr, 0, lane);
  const v16bf a1 = afrag_from_bf16(theta, FR, gr, 32, lane);

  // logits into LDS, with scale + distance mask
  for (int ct = 0; ct < 32; ++ct) {
    const int n0 = ct * 16;
    v8f acc = zero8();
    acc = wmma_bf16(a0, bfrag_from_lds(phils + (n0 + lr) * FR + 0  + hi * 16), acc);
    acc = wmma_bf16(a1, bfrag_from_lds(phils + (n0 + lr) * FR + 32 + hi * 16), acc);
    const int col = n0 + lr;
    const float ccx = cx[col], ccy = cy[col];
#pragma unroll
    for (int v = 0; v < 8; ++v) {
      const int row = rowbase + v + 8 * hi;  // frame-local
      const float dx = cx[row] - ccx, dy = cy[row] - ccy;
      float s = acc[v] * 0.125f;  // 1/sqrt(64)
      if (dx * dx + dy * dy > THR2) s = -INFINITY;
      mysim[(v + 8 * hi) * NN + col] = s;
    }
  }

  // per-row max & sum(exp) over the wave's 16 rows (wave32 reductions)
  for (int r = 0; r < 16; ++r) {
    const float* pr = mysim + r * NN;
    float m = -INFINITY;
#pragma unroll
    for (int j = 0; j < 16; ++j) m = fmaxf(m, pr[j * 32 + lane]);
#pragma unroll
    for (int o = 16; o > 0; o >>= 1) m = fmaxf(m, __shfl_xor(m, o, 32));
    float s = 0.f;
#pragma unroll
    for (int j = 0; j < 16; ++j) s += __expf(pr[j * 32 + lane] - m);
#pragma unroll
    for (int o = 16; o > 0; o >>= 1) s += __shfl_xor(s, o, 32);
    if (lane == 0) {
      rowm[wave * 16 + r]   = m;
      rowinv[wave * 16 + r] = 1.0f / s;
    }
  }

  // normalized writeout (coalesced float4 rows)
  for (int idx = lane; idx < 16 * (NN / 4); idx += 32) {
    const int r = idx >> 7;          // NN/4 = 128 float4 per row
    const int c4 = idx & 127;
    const float m = rowm[wave * 16 + r];
    const float inv = rowinv[wave * 16 + r];
    float4 v4 = *(const float4*)(mysim + r * NN + c4 * 4);
    v4.x = __expf(v4.x - m) * inv;
    v4.y = __expf(v4.y - m) * inv;
    v4.z = __expf(v4.z - m) * inv;
    v4.w = __expf(v4.w - m) * inv;
    *(float4*)(rel + (gr + r) * NN + c4 * 4) = v4;
  }
}

// ---- kernel 3: out = (rel @ X) @ W, fused -------------------------------
__global__ __launch_bounds__(128) void gcn_out_kernel(
    const float* __restrict__ rel, const float* __restrict__ X,
    const float* __restrict__ W, float* __restrict__ out) {
  __shared__ __bf16 Wls[FG][FG];      // Wls[n][k] = W[k][n]      32 KB
  __shared__ __bf16 Xls[FG][32];      // Xls[n][kk] = X[kb+kk][n]  8 KB
  __shared__ __bf16 Pls[4][16][FG];   // per-wave P tile          16 KB
  const int f = blockIdx.x >> 3, tile = blockIdx.x & 7;
  const int lane = threadIdx.x & 31, wave = threadIdx.x >> 5;
  const int lr = lane & 15, hi = lane >> 4;

  for (int i = threadIdx.x; i < FG * FG; i += 128) {
    const int k = i >> 7, n = i & 127;
    Wls[n][k] = (__bf16)W[i];
  }

  const int rowbase = tile * 64 + wave * 16;
  const long gr = (long)f * NN + rowbase;

  v8f acc[8];
#pragma unroll
  for (int nt = 0; nt < 8; ++nt) acc[nt] = zero8();

  for (int kc = 0; kc < 16; ++kc) {
    const int kb = kc * 32;
    // prefetch next k-slab of the relation rows (HBM-resident stream)
    if (kc + 1 < 16) {
      const float* pn = rel + (gr + lr) * (long)NN + kb + 32 + hi * 8;
      __builtin_prefetch(pn, 0, 1);
    }
    __syncthreads();
    for (int i = threadIdx.x; i < 32 * FG; i += 128) {
      const int kk = i >> 7, n = i & 127;
      Xls[n][kk] = (__bf16)X[((long)f * NN + kb + kk) * FG + n];
    }
    __syncthreads();
    const v16bf a = afrag_from_f32(rel, NN, gr, kb, lane);
#pragma unroll
    for (int nt = 0; nt < 8; ++nt) {
      acc[nt] = wmma_bf16(a, bfrag_from_lds(&Xls[nt * 16 + lr][hi * 16]), acc[nt]);
    }
  }

  // re-layout P (D-layout) -> A-fragments via LDS
#pragma unroll
  for (int nt = 0; nt < 8; ++nt)
#pragma unroll
    for (int v = 0; v < 8; ++v)
      Pls[wave][v + 8 * hi][nt * 16 + lr] = (__bf16)acc[nt][v];
  __syncthreads();

  v16bf pa[4];
#pragma unroll
  for (int kc = 0; kc < 4; ++kc) {
    const __bf16* p = &Pls[wave][lr][kc * 32 + hi * 16];
    union { v16bf v; uint4 q[2]; } u;
    u.q[0] = *(const uint4*)(p);
    u.q[1] = *(const uint4*)(p + 8);
    pa[kc] = u.v;
  }
#pragma unroll
  for (int nt = 0; nt < 8; ++nt) {
    v8f c2 = zero8();
#pragma unroll
    for (int kc = 0; kc < 4; ++kc) {
      c2 = wmma_bf16(pa[kc], bfrag_from_lds(&Wls[nt * 16 + lr][kc * 32 + hi * 16]), c2);
    }
#pragma unroll
    for (int v = 0; v < 8; ++v)
      out[(gr + v + 8 * hi) * FG + nt * 16 + lr] = c2[v];
  }
}

extern "C" void kernel_launch(void* const* d_in, const int* in_sizes, int n_in,
                              void* d_out, int out_size, void* d_ws, size_t ws_size,
                              hipStream_t stream) {
  (void)in_sizes; (void)n_in; (void)out_size; (void)ws_size;
  const float* X     = (const float*)d_in[0];  // [BT*NN, FG]
  const float* boxes = (const float*)d_in[1];  // [BT*NN, 4]
  const float* W     = (const float*)d_in[2];  // [FG, FG]
  const float* tw    = (const float*)d_in[3];  // [FR, FG]
  const float* tb    = (const float*)d_in[4];  // [FR]
  const float* pw    = (const float*)d_in[5];  // [FR, FG]
  const float* pb    = (const float*)d_in[6];  // [FR]

  float* out = (float*)d_out;                  // [BT*NN, FG]
  float* rel = out + (long)BT * NN * FG;       // [BT*NN, NN]

  __bf16* theta = (__bf16*)d_ws;               // 16 MB
  __bf16* phi   = theta + (long)BT * NN * FR;  // 16 MB

  gcn_proj_kernel<<<BT * (NN / 64), 128, 0, stream>>>(X, tw, tb, pw, pb, theta, phi);
  gcn_sim_softmax_kernel<<<BT * (NN / 64), 128, SM_BYTES, stream>>>(theta, phi, boxes, rel);
  gcn_out_kernel<<<BT * (NN / 64), 128, 0, stream>>>(rel, X, W, out);
}
